// PennyLaneBasicDQN_28432683499870
// MI455X (gfx1250) — compile-verified, matrix operand order checked by reading, not run on hardware
//
#include <hip/hip_runtime.h>
#include <hip/hip_bf16.h>

// ---------------------------------------------------------------------------
// PennyLane basic DQN forward on MI455X (gfx1250, wave32).
//
// 5 full-state HBM sweeps (state = 512 x 2^16 complex64 = 256 MB in d_ws):
//   P0 k_p0 : generate RY-embedded product state + L1 Rot q0..12   (W)
//   P1 k_pB : L1 Rot q13..15, CZ1, L2 Rot q10..15                  (R+W)
//   P2 k_p2 : L2 Rot q0..9,  CZ2, L3 Rot q0..12                    (R+W)
//   P3 k_pB : L3 Rot q13..15, CZ3, L4 Rot q10..15                  (R+W)
//   P4 k_p4 : L4 Rot q0..9 (CZ4 sign drops out of |amp|^2) +
//             per-qubit <Z> partial reduction                      (R)
//   k_zfin  : combine 8 tile-partials per sample -> z[512,16]
//   k_fc    : z @ fc_w^T + b via v_wmma_f32_16x16x32_f16
//
// ~2.05 GB HBM traffic -> ~88 us at 23.3 TB/s.  Inside each sweep, threads
// keep 32 amps in registers (5 register-local qubits per phase) and use
// swizzled LDS only to re-map which qubits are register-local: 3 LDS round
// trips per sweep instead of 13 gate-wise R+W passes.
// ---------------------------------------------------------------------------

#define NQ      16
#define NSTATE  65536
#define BATCH   512
#define NACT    6
#define TILE    8192          // 2^13 amps per workgroup tile
#define WSTRIDE 48            // floats per layer of weights (16*3)

typedef __attribute__((ext_vector_type(16))) _Float16 v16h;
typedef __attribute__((ext_vector_type(8)))  float    v8f;

// XOR swizzle: kills LDS bank conflicts in the mapping-exchange phases.
__device__ __forceinline__ int SW(int L) { return L ^ ((L >> 5) & 31); }

// PennyLane Rot(phi,theta,omega) = RZ(omega) RY(theta) RZ(phi):
// m = {m00r,m00i, m01r,m01i, m10r,m10i, m11r,m11i}
__device__ __forceinline__ void rot_mat(const float* w, float* m) {
    float c, s, ca, sa, cb, sb;
    const float phi = w[0], th = w[1], om = w[2];
    __sincosf(0.5f * th, &s, &c);
    __sincosf(0.5f * (phi + om), &sa, &ca);
    __sincosf(0.5f * (phi - om), &sb, &cb);
    m[0] =  c * ca;  m[1] = -c * sa;
    m[2] = -s * cb;  m[3] = -s * sb;
    m[4] =  s * cb;  m[5] = -s * sb;
    m[6] =  c * ca;  m[7] =  c * sa;
}

// CZ chain sign on (0,1)..(14,15)
__device__ __forceinline__ float cz_sign(int n) {
    return (__popc(n & (n >> 1) & 0x7FFF) & 1) ? -1.0f : 1.0f;
}

// Apply a 2x2 complex gate on register bit RB of 32 register-resident amps.
template<int RB>
__device__ __forceinline__ void gate_regs(float (&xr)[32], float (&xi)[32],
                                          const float* gmem) {
    float g[8];
    #pragma unroll
    for (int i = 0; i < 8; ++i) g[i] = gmem[i];
    #pragma unroll
    for (int p = 0; p < 16; ++p) {
        const int j0 = ((p >> RB) << (RB + 1)) | (p & ((1 << RB) - 1));
        const int j1 = j0 | (1 << RB);
        const float ar = xr[j0], ai = xi[j0], br = xr[j1], bi = xi[j1];
        xr[j0] = g[0] * ar - g[1] * ai + g[2] * br - g[3] * bi;
        xi[j0] = g[0] * ai + g[1] * ar + g[2] * bi + g[3] * br;
        xr[j1] = g[4] * ar - g[5] * ai + g[6] * br - g[7] * bi;
        xi[j1] = g[4] * ai + g[5] * ar + g[6] * bi + g[7] * br;
    }
}

// Register mappings over A-tile canonical local index L = n[12:0]:
//   A0: reg j = n[4:0],            tid = n[12:5]
//   A1: reg j = n[9:5],            tid = {n[12:10], n[4:0]}
//   A2: reg j = {n[12:10], n[1:0]},tid = n[9:2]
#define A0_ADDR(tid, j) (((tid) << 5) | (j))
#define A1_ADDR(tid, j) ((((tid) >> 5) << 10) | ((j) << 5) | ((tid) & 31))
#define A2_ADDR(tid, j) ((((j) >> 2) << 10) | ((tid) << 2) | ((j) & 3))

#define XCHG_WR(ADDR)                                              \
    _Pragma("unroll")                                              \
    for (int j = 0; j < 32; ++j) {                                 \
        const int P = SW(ADDR(tid, j));                            \
        sre[P] = xr[j]; sim[P] = xi[j];                            \
    }
#define XCHG_RD(ADDR)                                              \
    _Pragma("unroll")                                              \
    for (int j = 0; j < 32; ++j) {                                 \
        const int P = SW(ADDR(tid, j));                            \
        xr[j] = sre[P]; xi[j] = sim[P];                            \
    }

// -------- P0: generate embedded state + L1 Rot q0..12, write-only ----------
__global__ void k_p0(const float* __restrict__ x,
                     const float* __restrict__ w1,
                     float2* __restrict__ state) {
    __shared__ float sre[TILE], sim[TILE];
    __shared__ float gm[13][8];
    __shared__ float cq[NQ], sq[NQ];

    const int tid = threadIdx.x;
    const int b = blockIdx.x >> 3, tt = blockIdx.x & 7;   // tt = n[15:13]

    if (tid < NQ) {
        float sn, cs;
        __sincosf(0.5f * x[b * NQ + tid], &sn, &cs);
        cq[tid] = cs; sq[tid] = sn;
    }
    if (tid < 13) rot_mat(w1 + 3 * tid, gm[tid]);
    __syncthreads();

    float xr[32], xi[32];
    {   // product state, mapping A0 (thread's 32 amps are contiguous)
        const int nhi = (tt << 13) | (tid << 5);
        float base = 1.0f;
        #pragma unroll
        for (int i = 5; i < NQ; ++i) base *= ((nhi >> i) & 1) ? sq[i] : cq[i];
        #pragma unroll
        for (int j = 0; j < 32; ++j) {
            float p = base;
            #pragma unroll
            for (int i = 0; i < 5; ++i) p *= ((j >> i) & 1) ? sq[i] : cq[i];
            xr[j] = p; xi[j] = 0.0f;
        }
    }
    gate_regs<0>(xr, xi, gm[0]); gate_regs<1>(xr, xi, gm[1]);
    gate_regs<2>(xr, xi, gm[2]); gate_regs<3>(xr, xi, gm[3]);
    gate_regs<4>(xr, xi, gm[4]);

    XCHG_WR(A0_ADDR); __syncthreads(); XCHG_RD(A1_ADDR);
    gate_regs<0>(xr, xi, gm[5]); gate_regs<1>(xr, xi, gm[6]);
    gate_regs<2>(xr, xi, gm[7]); gate_regs<3>(xr, xi, gm[8]);
    gate_regs<4>(xr, xi, gm[9]);

    XCHG_WR(A1_ADDR); __syncthreads(); XCHG_RD(A2_ADDR);
    gate_regs<2>(xr, xi, gm[10]); gate_regs<3>(xr, xi, gm[11]);
    gate_regs<4>(xr, xi, gm[12]);
    XCHG_WR(A2_ADDR); __syncthreads();

    float4* st4 = (float4*)(state + ((size_t)b << NQ));
    #pragma unroll
    for (int k = 0; k < 16; ++k) {
        const int i2 = tid + (k << 8), L0 = i2 << 1;
        st4[(tt << 12) + i2] = make_float4(sre[SW(L0)], sim[SW(L0)],
                                           sre[SW(L0 | 1)], sim[SW(L0 | 1)]);
    }
}

// -------- P1/P3: B-tile (bits {10..15} local, n[9:7] = tile id) -------------
// applies: layer-l Rot q13..15, CZ(l), layer-(l+1) Rot q10..15
__global__ void k_pB(const float* __restrict__ wl,
                     const float* __restrict__ wn,
                     float2* __restrict__ state) {
    __shared__ float sre[TILE], sim[TILE];
    __shared__ float gmH[3][8], gmN[6][8];

    const int tid = threadIdx.x;
    const int b = blockIdx.x >> 3, f3 = blockIdx.x & 7;   // f3 = n[9:7]

    if (tid < 3)      rot_mat(wl + 3 * (13 + tid), gmH[tid]);
    else if (tid < 9) rot_mat(wn + 3 * (10 + (tid - 3)), gmN[tid - 3]);

    float4* st4 = (float4*)(state + ((size_t)b << NQ));
    // canonical B-tile local index: L[12:7] = n[15:10], L[6:0] = n[6:0]
    #pragma unroll
    for (int k = 0; k < 16; ++k) {
        const int i2 = tid + (k << 8), hi6 = i2 >> 6, lo6 = i2 & 63;
        const float4 v = st4[(hi6 << 9) | (f3 << 6) | lo6];
        const int L0 = (hi6 << 7) | (lo6 << 1);
        sre[SW(L0)] = v.x;     sim[SW(L0)] = v.y;
        sre[SW(L0 | 1)] = v.z; sim[SW(L0 | 1)] = v.w;
    }
    __syncthreads();

    float xr[32], xi[32];
    // B0: reg j = L[12:8] = n[15:11]; tid = {n[10], n[6:0]}
    #pragma unroll
    for (int j = 0; j < 32; ++j) {
        const int P = SW((j << 8) | tid);
        xr[j] = sre[P]; xi[j] = sim[P];
    }
    gate_regs<2>(xr, xi, gmH[0]);      // l: q13
    gate_regs<3>(xr, xi, gmH[1]);      // l: q14
    gate_regs<4>(xr, xi, gmH[2]);      // l: q15
    #pragma unroll
    for (int j = 0; j < 32; ++j) {     // CZ(l)
        const int n = (j << 11) | ((tid & 128) << 3) | (f3 << 7) | (tid & 127);
        const float s = cz_sign(n);
        xr[j] *= s; xi[j] *= s;
    }
    gate_regs<0>(xr, xi, gmN[1]);      // l+1: q11
    gate_regs<1>(xr, xi, gmN[2]);      // l+1: q12
    gate_regs<2>(xr, xi, gmN[3]);      // l+1: q13
    gate_regs<3>(xr, xi, gmN[4]);      // l+1: q14
    gate_regs<4>(xr, xi, gmN[5]);      // l+1: q15
    #pragma unroll
    for (int j = 0; j < 32; ++j) {
        const int P = SW((j << 8) | tid);
        sre[P] = xr[j]; sim[P] = xi[j];
    }
    __syncthreads();

    // B1: reg j = {n[10], n[3:0]}; tid = {n[15:11], n[6:4]} = (h<<3)|m
    const int h = tid >> 3, m = tid & 7;
    #pragma unroll
    for (int j = 0; j < 32; ++j) {
        const int P = SW((h << 8) | ((j >> 4) << 7) | (m << 4) | (j & 15));
        xr[j] = sre[P]; xi[j] = sim[P];
    }
    gate_regs<4>(xr, xi, gmN[0]);      // l+1: q10
    #pragma unroll
    for (int j = 0; j < 32; ++j) {
        const int P = SW((h << 8) | ((j >> 4) << 7) | (m << 4) | (j & 15));
        sre[P] = xr[j]; sim[P] = xi[j];
    }
    __syncthreads();

    #pragma unroll
    for (int k = 0; k < 16; ++k) {
        const int i2 = tid + (k << 8), hi6 = i2 >> 6, lo6 = i2 & 63;
        const int L0 = (hi6 << 7) | (lo6 << 1);
        st4[(hi6 << 9) | (f3 << 6) | lo6] =
            make_float4(sre[SW(L0)], sim[SW(L0)], sre[SW(L0 | 1)], sim[SW(L0 | 1)]);
    }
}

// -------- P2: A-tile: L2 q0..9, CZ2, L3 q0..12 ------------------------------
__global__ void k_p2(const float* __restrict__ wA,   // layer l   (q0..9)
                     const float* __restrict__ wB,   // layer l+1 (q0..12)
                     float2* __restrict__ state) {
    __shared__ float sre[TILE], sim[TILE];
    __shared__ float gmA[10][8], gmB[13][8];

    const int tid = threadIdx.x;
    const int b = blockIdx.x >> 3, tt = blockIdx.x & 7;

    if (tid < 10)      rot_mat(wA + 3 * tid, gmA[tid]);
    else if (tid < 23) rot_mat(wB + 3 * (tid - 10), gmB[tid - 10]);

    float4* st4 = (float4*)(state + ((size_t)b << NQ));
    float xr[32], xi[32];
    #pragma unroll
    for (int q = 0; q < 16; ++q) {      // thread's 32 contiguous amps (A0)
        const float4 v = st4[(tt << 12) + (tid << 4) + q];
        xr[2 * q] = v.x; xi[2 * q] = v.y; xr[2 * q + 1] = v.z; xi[2 * q + 1] = v.w;
    }
    __syncthreads();                    // gate matrices ready

    gate_regs<0>(xr, xi, gmA[0]); gate_regs<1>(xr, xi, gmA[1]);  // L_l q0..4
    gate_regs<2>(xr, xi, gmA[2]); gate_regs<3>(xr, xi, gmA[3]);
    gate_regs<4>(xr, xi, gmA[4]);

    XCHG_WR(A0_ADDR); __syncthreads(); XCHG_RD(A1_ADDR);
    gate_regs<0>(xr, xi, gmA[5]); gate_regs<1>(xr, xi, gmA[6]);  // L_l q5..9
    gate_regs<2>(xr, xi, gmA[7]); gate_regs<3>(xr, xi, gmA[8]);
    gate_regs<4>(xr, xi, gmA[9]);
    #pragma unroll
    for (int j = 0; j < 32; ++j) {      // CZ(l) in A1 mapping
        const int n = (tt << 13) | ((tid >> 5) << 10) | (j << 5) | (tid & 31);
        const float s = cz_sign(n);
        xr[j] *= s; xi[j] *= s;
    }
    gate_regs<0>(xr, xi, gmB[5]); gate_regs<1>(xr, xi, gmB[6]);  // L_{l+1} q5..9
    gate_regs<2>(xr, xi, gmB[7]); gate_regs<3>(xr, xi, gmB[8]);
    gate_regs<4>(xr, xi, gmB[9]);

    XCHG_WR(A1_ADDR); __syncthreads(); XCHG_RD(A2_ADDR);
    gate_regs<2>(xr, xi, gmB[10]); gate_regs<3>(xr, xi, gmB[11]);
    gate_regs<4>(xr, xi, gmB[12]);                               // q10..12

    XCHG_WR(A2_ADDR); __syncthreads(); XCHG_RD(A0_ADDR);
    gate_regs<0>(xr, xi, gmB[0]); gate_regs<1>(xr, xi, gmB[1]);  // L_{l+1} q0..4
    gate_regs<2>(xr, xi, gmB[2]); gate_regs<3>(xr, xi, gmB[3]);
    gate_regs<4>(xr, xi, gmB[4]);

    #pragma unroll
    for (int q = 0; q < 16; ++q)
        st4[(tt << 12) + (tid << 4) + q] =
            make_float4(xr[2 * q], xi[2 * q], xr[2 * q + 1], xi[2 * q + 1]);
}

// -------- P4: L4 q0..9 + per-qubit <Z> partial reduction (read-only) --------
__global__ void k_p4(const float* __restrict__ w4,
                     const float2* __restrict__ state,
                     float* __restrict__ zp) {      // [4096][17] partials
    __shared__ float sre[TILE], sim[TILE];
    __shared__ float gm[10][8];
    __shared__ float smem[256];

    const int tid = threadIdx.x;
    const int b = blockIdx.x >> 3, tt = blockIdx.x & 7;

    if (tid < 10) rot_mat(w4 + 3 * tid, gm[tid]);

    const float4* st4 = (const float4*)(state + ((size_t)b << NQ));
    float xr[32], xi[32];
    #pragma unroll
    for (int q = 0; q < 16; ++q) {
        const float4 v = st4[(tt << 12) + (tid << 4) + q];
        xr[2 * q] = v.x; xi[2 * q] = v.y; xr[2 * q + 1] = v.z; xi[2 * q + 1] = v.w;
    }
    __syncthreads();

    gate_regs<0>(xr, xi, gm[0]); gate_regs<1>(xr, xi, gm[1]);
    gate_regs<2>(xr, xi, gm[2]); gate_regs<3>(xr, xi, gm[3]);
    gate_regs<4>(xr, xi, gm[4]);
    XCHG_WR(A0_ADDR); __syncthreads(); XCHG_RD(A1_ADDR);
    gate_regs<0>(xr, xi, gm[5]); gate_regs<1>(xr, xi, gm[6]);
    gate_regs<2>(xr, xi, gm[7]); gate_regs<3>(xr, xi, gm[8]);
    gate_regs<4>(xr, xi, gm[9]);

    float acc[NQ], tot = 0.0f;
    #pragma unroll
    for (int i = 0; i < NQ; ++i) acc[i] = 0.0f;
    #pragma unroll
    for (int j = 0; j < 32; ++j) {
        const int n = (tt << 13) | ((tid >> 5) << 10) | (j << 5) | (tid & 31);
        const float p = xr[j] * xr[j] + xi[j] * xi[j];
        tot += p;
        #pragma unroll
        for (int i = 0; i < NQ; ++i) acc[i] += ((n >> i) & 1) ? p : 0.0f;
    }
    __syncthreads();
    for (int i = 0; i < 17; ++i) {
        smem[tid] = (i < NQ) ? acc[i] : tot;
        __syncthreads();
        for (int s = 128; s > 0; s >>= 1) {
            if (tid < s) smem[tid] += smem[tid + s];
            __syncthreads();
        }
        if (tid == 0) zp[blockIdx.x * 17 + i] = smem[0];
        __syncthreads();
    }
}

// -------- combine tile partials -> z[512,16] --------------------------------
__global__ void k_zfin(const float* __restrict__ zp, float* __restrict__ z) {
    const int b = blockIdx.x * blockDim.x + threadIdx.x;
    if (b >= BATCH) return;
    float S[NQ], tot = 0.0f;
    #pragma unroll
    for (int i = 0; i < NQ; ++i) S[i] = 0.0f;
    for (int t = 0; t < 8; ++t) {
        const float* p = zp + (b * 8 + t) * 17;
        #pragma unroll
        for (int i = 0; i < NQ; ++i) S[i] += p[i];
        tot += p[16];
    }
    #pragma unroll
    for (int i = 0; i < NQ; ++i) z[b * NQ + i] = tot - 2.0f * S[i];
}

// ---------- FC: z[512,16] @ fc_w^T[16,6] + b via WMMA f32_16x16x32_f16 ------
__global__ void k_fc(const float* __restrict__ z,
                     const float* __restrict__ fc_w,
                     const float* __restrict__ fc_b,
                     float* __restrict__ out) {
    const int lane = threadIdx.x;          // one wave per block
    const int tile = blockIdx.x;           // 32 row tiles of 16
    const int half = lane >> 4;
    const int l16  = lane & 15;

    v16h a;                                // A: 16x32 f16 (K 16..31 zero pad)
    const int row = tile * 16 + l16;
    #pragma unroll
    for (int j = 0; j < 8; ++j)
        a[j] = (_Float16)z[row * NQ + (half ? 8 + j : j)];
    #pragma unroll
    for (int j = 8; j < 16; ++j) a[j] = (_Float16)0.0f;

    v16h bb;                               // B: 32x16 f16 (N 6..15 zero pad)
    #pragma unroll
    for (int j = 0; j < 8; ++j) {
        const int kk = (half ? 8 + j : j);
        bb[j] = (_Float16)((l16 < NACT) ? fc_w[l16 * NQ + kk] : 0.0f);
    }
    #pragma unroll
    for (int j = 8; j < 16; ++j) bb[j] = (_Float16)0.0f;

    v8f c = {};
    c = __builtin_amdgcn_wmma_f32_16x16x32_f16(
            false, a, false, bb, (short)0, c, false, false);

    const int col = l16;
    if (col < NACT) {
        const float bias = fc_b[col];
        #pragma unroll
        for (int r = 0; r < 8; ++r) {
            const int M = r + half * 8;
            out[(tile * 16 + M) * NACT + col] = c[r] + bias;
        }
    }
}

// ---------------------------------------------------------------------------
extern "C" void kernel_launch(void* const* d_in, const int* in_sizes, int n_in,
                              void* d_out, int out_size, void* d_ws, size_t ws_size,
                              hipStream_t stream) {
    const float* x       = (const float*)d_in[0];   // [512,16]
    const float* W       = (const float*)d_in[1];   // [4,16,3]
    const float* fc_w    = (const float*)d_in[2];   // [6,16]
    const float* fc_b    = (const float*)d_in[3];   // [6]
    float* out = (float*)d_out;                     // [512,6]

    const size_t STATE_BYTES = (size_t)BATCH * NSTATE * sizeof(float2); // 256 MB
    float2* state = (float2*)d_ws;
    float*  zp    = (float*)((char*)d_ws + STATE_BYTES);   // [4096][17]
    float*  z     = zp + 4096 * 17;                        // [512][16]

    const int NB = BATCH * 8;   // 4096 workgroups, 256 threads each

    k_p0<<<NB, 256, 0, stream>>>(x, W + 0 * WSTRIDE, state);
    k_pB<<<NB, 256, 0, stream>>>(W + 0 * WSTRIDE, W + 1 * WSTRIDE, state);
    k_p2<<<NB, 256, 0, stream>>>(W + 1 * WSTRIDE, W + 2 * WSTRIDE, state);
    k_pB<<<NB, 256, 0, stream>>>(W + 2 * WSTRIDE, W + 3 * WSTRIDE, state);
    k_p4<<<NB, 256, 0, stream>>>(W + 3 * WSTRIDE, state, zp);

    k_zfin<<<4, 128, 0, stream>>>(zp, z);
    k_fc<<<BATCH / 16, 32, 0, stream>>>(z, fc_w, fc_b, out);
}
